// PegasusXGlobalLocalAttention_71107478552592
// MI455X (gfx1250) — matrix operand order repaired and checked by reading, not compiled, not run on hardware
//
#include <hip/hip_runtime.h>
#include <hip/hip_bf16.h>
#include <math.h>

// ---------------- CDNA5 WMMA types ----------------
typedef __attribute__((ext_vector_type(16))) __bf16 v16bf;
typedef __attribute__((ext_vector_type(8)))  __bf16 v8bf;
typedef __attribute__((ext_vector_type(8)))  float  v8f;
typedef __attribute__((ext_vector_type(4)))  unsigned int v4u;
typedef __attribute__((ext_vector_type(8)))  int v8i;
typedef __attribute__((ext_vector_type(4)))  int v4i;

#define D_MODEL 1024
#define N_HEADS 16
#define D_HEAD  64
#define BLK     64
#define G_LEN   64
#define SEQ     8192
#define N_BLKS  (SEQ / BLK)
#define SCALE_F 0.125f

static __device__ __forceinline__ __bf16 f2bf(float f) { return (__bf16)f; }

static __device__ __forceinline__ v8f wmma_bf16(v16bf a, v16bf b, v8f c) {
  // D = A(16x32 bf16) x B(32x16 bf16) + C(16x16 f32)
  return __builtin_amdgcn_wmma_f32_16x16x32_bf16(false, a, false, b, (short)0, c,
                                                 false, false);
}

// two aligned 16B loads + register concat -> one WMMA operand
static __device__ __forceinline__ v8bf ldv8(const __bf16* p) {
  return *(const v8bf*)p;
}
static __device__ __forceinline__ v16bf cat8(v8bf lo, v8bf hi) {
  return __builtin_shufflevector(lo, hi, 0, 1, 2, 3, 4, 5, 6, 7,
                                 8, 9, 10, 11, 12, 13, 14, 15);
}

// ---------------- CDNA5 async global->LDS staging (ASYNCcnt path) ----------------
#define USE_ASYNC_LDS 1

static __device__ __forceinline__ unsigned lds_addr_of(const void* p) {
  // generic -> LDS address space -> 32-bit LDS byte address
  return (unsigned)(unsigned long long)(__attribute__((address_space(3))) const void*)p;
}

static __device__ __forceinline__ void stage16(const __bf16* g, __bf16* l) {
#if USE_ASYNC_LDS
  const unsigned lds = lds_addr_of(l);
  const unsigned long long ga = (unsigned long long)g;
  asm volatile("global_load_async_to_lds_b128 %0, %1, off"
               :: "v"(lds), "v"(ga) : "memory");
#else
  *(v8bf*)l = *(const v8bf*)g;
#endif
}
static __device__ __forceinline__ void stage_join() {
#if USE_ASYNC_LDS
  asm volatile("s_wait_asynccnt 0x0" ::: "memory");
#endif
}

// ---------------- CDNA5 Tensor Data Mover: 64x64 bf16 tile -> LDS ----------------
// D# per ISA ch.8: 2D tile, data_size=2B, tile_dim0=64 (contiguous feats),
// tile_dim1=64 (rows, stride=1024 elems), LDS padding 4 DW after every 32 DW
// so the landed row stride is 144B (= LDK elements).
#if __has_builtin(__builtin_amdgcn_tensor_load_to_lds)
#define HAS_TDM 1
#else
#define HAS_TDM 0
#endif

static __device__ __forceinline__ void tdm_load_tile(const __bf16* gsrc,
                                                     unsigned lds_addr) {
#if HAS_TDM
  const unsigned long long ga = (unsigned long long)gsrc;
  v4u g0;
  g0[0] = 1u;                                          // count=1 (valid user D#)
  g0[1] = lds_addr;                                    // lds_addr [63:32]
  g0[2] = (unsigned)(ga & 0xffffffffu);                // global_addr low
  g0[3] = (unsigned)((ga >> 32) & 0x01ffffffu)         // global_addr [56:32]
          | (2u << 30);                                // type=2 ("image")
  v8i g1;
  g1[0] = (int)((1u << 16)       // data_size = 2 bytes
                | (1u << 20)     // pad_enable
                | (4u << 22)     // pad_interval: 32 DWORDs (one 128B row)
                | (3u << 25));   // pad_amount: 4 DWORDs (16B -> 144B stride)
  g1[1] = (int)(1024u << 16);    // tensor_dim0 = 1024 (low 16 into bits 63:48)
  g1[2] = 0;                     // tensor_dim0 hi / tensor_dim1 lo
  g1[3] = (int)(0x10u | (64u << 16));  // tensor_dim1 = 1<<20, tile_dim0 = 64
  g1[4] = 64;                    // tile_dim1 = 64, tile_dim2 = 0
  g1[5] = 1024;                  // tensor_dim0_stride = 1024 elements
  g1[6] = 0;                     // stride0 hi / stride1 lo
  g1[7] = 0;                     // stride1 hi
  v4i zero4 = {};
  v8i zero8 = {};
  // 6-arg toolchain variant: (g0, g1, g2, g3, g4, cpol)
  __builtin_amdgcn_tensor_load_to_lds(g0, g1, zero4, zero4, zero8, 0);
  __builtin_amdgcn_s_wait_tensorcnt(0);
#else
  (void)gsrc; (void)lds_addr;
#endif
}

// ---------------- weight transpose + fp32->bf16:  Wt[n][k] = bf16(W[k][n]) ----
__global__ __launch_bounds__(256)
void transpose_w_kernel(const float* __restrict__ W, __bf16* __restrict__ Wt) {
  __shared__ __bf16 lT[32 * 33];
  const int k0 = blockIdx.x * 32, n0 = blockIdx.y * 32;
  const int c = threadIdx.x & 31, r = threadIdx.x >> 5;  // 32 x 8
#pragma unroll
  for (int i = 0; i < 4; ++i) {
    const int kl = r + i * 8;
    lT[kl * 33 + c] = f2bf(W[(size_t)(k0 + kl) * D_MODEL + n0 + c]);
  }
  __syncthreads();
#pragma unroll
  for (int i = 0; i < 4; ++i) {
    const int nl = r + i * 8;
    Wt[(size_t)(n0 + nl) * D_MODEL + k0 + c] = lT[c * 33 + nl];
  }
}

// ---------------- tiled WMMA GEMM:  Y = (X @ W + bias) * scale ----------------
// X: [rows,1024] (fp32 Xf or bf16 Xb), Wt: [N][K] bf16 (transposed weights).
// grid = (rows/64, 1024/64), block = 256 = 8 waves.
#define GEMM_LDA 40   // 16B-aligned row stride (80B), 20-bank rotation
#define GEMM_LDB 40

template <bool HAS_F32>
__global__ __launch_bounds__(256)
void gemm_bias_kernel(const float* __restrict__ Xf, const __bf16* __restrict__ Xb,
                      const __bf16* __restrict__ Wt, const float* __restrict__ bias,
                      float scale, __bf16* __restrict__ Yb, float* __restrict__ Yf) {
  __shared__ __attribute__((aligned(16))) __bf16 lA [BLK * GEMM_LDA];  // 64m x 32k
  __shared__ __attribute__((aligned(16))) __bf16 lBt[BLK * GEMM_LDB];  // 64n x 32k

  const int tid  = threadIdx.x;
  const int lane = tid & 31;
  const int wid  = tid >> 5;
  const int half = lane >> 4;
  const int l16  = lane & 15;

  const int m0 = blockIdx.x * BLK;
  const int n0 = blockIdx.y * BLK;
  const int mt = wid >> 1;
  const int nh = wid & 1;

  v8f c0 = {}; v8f c1 = {};

  const int a_m = tid >> 2;            // 64 rows, 4 thr/row, 8 elems (16B) each
  const int a_k = (tid & 3) * 8;

  for (int kt = 0; kt < D_MODEL / 32; ++kt) {
    const int k0 = kt * 32;
    // ---- stage A tile ----
    if (HAS_F32) {
      const float* p = Xf + (size_t)(m0 + a_m) * D_MODEL + k0 + a_k;
      v8bf t;
#pragma unroll
      for (int e = 0; e < 8; ++e) t[e] = f2bf(p[e]);
      *(v8bf*)&lA[a_m * GEMM_LDA + a_k] = t;
    } else {
      stage16(Xb + (size_t)(m0 + a_m) * D_MODEL + k0 + a_k,
              &lA[a_m * GEMM_LDA + a_k]);
    }
    // ---- stage B tile (already transposed: [n][k]) ----
    stage16(Wt + (size_t)(n0 + a_m) * D_MODEL + k0 + a_k,
            &lBt[a_m * GEMM_LDB + a_k]);
    // prefetch next tiles (gfx1250 global_prefetch_b8)
    if (kt + 1 < D_MODEL / 32) {
      if (HAS_F32) __builtin_prefetch(Xf + (size_t)(m0 + a_m) * D_MODEL + k0 + 32 + a_k, 0, 1);
      else         __builtin_prefetch(Xb + (size_t)(m0 + a_m) * D_MODEL + k0 + 32 + a_k, 0, 1);
      __builtin_prefetch(Wt + (size_t)(n0 + a_m) * D_MODEL + k0 + 32 + a_k, 0, 1);
    }
    stage_join();
    __syncthreads();

    // ---- fragments: two 16B LDS loads each ----
    const __bf16* pa = &lA[(mt * 16 + l16) * GEMM_LDA + (half ? 8 : 0)];
    v16bf a = cat8(ldv8(pa), ldv8(pa + 16));

    const __bf16* pb0 = &lBt[((nh * 2) * 16 + l16) * GEMM_LDB + (half ? 8 : 0)];
    const __bf16* pb1 = &lBt[((nh * 2 + 1) * 16 + l16) * GEMM_LDB + (half ? 8 : 0)];
    v16bf b0 = cat8(ldv8(pb0), ldv8(pb0 + 16));
    v16bf b1 = cat8(ldv8(pb1), ldv8(pb1 + 16));

    c0 = wmma_bf16(a, b0, c0);
    c1 = wmma_bf16(a, b1, c1);
    __syncthreads();
  }

  // ---- epilogue: bias + scale ----
#pragma unroll
  for (int i = 0; i < 8; ++i) {
    const int m  = m0 + mt * 16 + i + (half ? 8 : 0);
    const int nA = n0 + (nh * 2) * 16 + l16;
    const int nB = n0 + (nh * 2 + 1) * 16 + l16;
    const float vA = (c0[i] + bias[nA]) * scale;
    const float vB = (c1[i] + bias[nB]) * scale;
    if (Yb) {
      Yb[(size_t)m * D_MODEL + nA] = f2bf(vA);
      Yb[(size_t)m * D_MODEL + nB] = f2bf(vB);
    }
    if (Yf) {
      Yf[(size_t)m * D_MODEL + nA] = vA;
      Yf[(size_t)m * D_MODEL + nB] = vB;
    }
  }
}

// ---------------- unified streaming attention (flash-style, online softmax) ----
// grid = (n_q_blocks, H, B), block = 256.  Chunk 0 = global K/V, chunks >=1 =
// local K/V blocks (local_mode: only own block; global mode: all 128 blocks).
#define LDK   72   // 144B row stride: 16B-aligned, matches TDM pad descriptor
#define LDS_S 66
#define LDACC 66

__global__ __launch_bounds__(256)
void attn_kernel(const __bf16* __restrict__ Qp, const __bf16* __restrict__ Kp,
                 const __bf16* __restrict__ Vp, const __bf16* __restrict__ Gk,
                 const __bf16* __restrict__ Gv, const float* __restrict__ mask,
                 __bf16* __restrict__ Out, int n_chunks, float score_scale,
                 int local_mode) {
  __shared__ __attribute__((aligned(16))) __bf16 lK [BLK    * LDK];  // [key][feat]
  __shared__ __attribute__((aligned(16))) __bf16 lVt[D_HEAD * LDK];  // [feat][key]
  __shared__ __attribute__((aligned(16))) __bf16 lP [BLK    * LDK];  // probs bf16
  __shared__ float lS  [BLK * LDS_S];
  __shared__ float lAcc[BLK * LDACC];
  __shared__ float rowMax[BLK], rowSum[BLK], rowFac[BLK], lMask[BLK];

  const int tid  = threadIdx.x;
  const int lane = tid & 31;
  const int wid  = tid >> 5;
  const int half = lane >> 4;
  const int l16  = lane & 15;
  const int h  = blockIdx.y;
  const int b  = blockIdx.z;
  const int qb = blockIdx.x;

  const size_t q_row0 = local_mode ? ((size_t)b * SEQ + (size_t)qb * BLK)
                                   : ((size_t)b * G_LEN);
  const int hcol = h * D_HEAD;

  if (tid < BLK) { rowMax[tid] = -1.0e30f; rowSum[tid] = 0.0f; }
  for (int i = tid; i < BLK * LDACC; i += 256) lAcc[i] = 0.0f;
  __syncthreads();

  const int mt = wid >> 1;
  const int nh = wid & 1;

  // ---- Q A-fragments held in VGPRs across all key chunks ----
  v16bf aQ[2];
  {
    const __bf16* qrow = Qp + (q_row0 + (size_t)(mt * 16 + l16)) * D_MODEL + hcol;
#pragma unroll
    for (int kt = 0; kt < 2; ++kt) {
      const int kb = kt * 32 + (half ? 8 : 0);
      aQ[kt] = cat8(*(const v8bf*)(qrow + kb), *(const v8bf*)(qrow + kb + 16));
    }
  }

  const int s_row = tid >> 2;        // staging: 64 rows, 4 thr/row
  const int s_f   = (tid & 3) * 16;  // 16 feats (32B) each

  for (int c = 0; c < n_chunks; ++c) {
    // ---- stage K (TDM / async) and V (transposed, manual) chunk ----
    const __bf16* ksrc; const __bf16* vsrc; size_t key_row0; int kb_idx = 0;
    if (c == 0) { key_row0 = (size_t)b * G_LEN; ksrc = Gk; vsrc = Gv; }
    else {
      kb_idx   = local_mode ? qb : (c - 1);
      key_row0 = (size_t)b * SEQ + (size_t)kb_idx * BLK;
      ksrc = Kp; vsrc = Vp;
    }
#if HAS_TDM
    if (wid == 0) {
      // one Tensor-DMA op moves the whole 64x64 bf16 K tile, padded to LDK
      tdm_load_tile(ksrc + key_row0 * D_MODEL + hcol, lds_addr_of(lK));
    }
#else
    {
      const __bf16* pk = ksrc + (key_row0 + s_row) * D_MODEL + hcol + s_f;
      stage16(pk,     &lK[s_row * LDK + s_f]);
      stage16(pk + 8, &lK[s_row * LDK + s_f + 8]);
    }
#endif
    {
      const __bf16* pv = vsrc + (key_row0 + s_row) * D_MODEL + hcol + s_f;
      v8bf r0 = *(const v8bf*)pv;
      v8bf r1 = *(const v8bf*)(pv + 8);
#pragma unroll
      for (int e = 0; e < 8; ++e) lVt[(s_f + e) * LDK + s_row] = r0[e];
#pragma unroll
      for (int e = 0; e < 8; ++e) lVt[(s_f + 8 + e) * LDK + s_row] = r1[e];
    }
    if (tid < BLK)
      lMask[tid] = (c == 0) ? 0.0f
                            : mask[(size_t)b * SEQ + (size_t)kb_idx * BLK + tid];
    stage_join();
    __syncthreads();

    // ---- scores = Q x K^T ----
    {
      v8f c0 = {}; v8f c1 = {};
      const int keyA = (nh * 2) * 16 + l16;
      const int keyB = (nh * 2 + 1) * 16 + l16;
#pragma unroll
      for (int kt = 0; kt < 2; ++kt) {
        const int kk = kt * 32 + (half ? 8 : 0);
        const __bf16* p0 = &lK[keyA * LDK + kk];
        const __bf16* p1 = &lK[keyB * LDK + kk];
        v16bf b0 = cat8(ldv8(p0), ldv8(p0 + 16));
        v16bf b1 = cat8(ldv8(p1), ldv8(p1 + 16));
        c0 = wmma_bf16(aQ[kt], b0, c0);
        c1 = wmma_bf16(aQ[kt], b1, c1);
      }
#pragma unroll
      for (int i = 0; i < 8; ++i) {
        const int m = mt * 16 + i + (half ? 8 : 0);
        lS[m * LDS_S + keyA] = c0[i] * score_scale + lMask[keyA];
        lS[m * LDS_S + keyB] = c1[i] * score_scale + lMask[keyB];
      }
    }
    __syncthreads();

    // ---- online softmax update (one thread per query row) ----
    if (tid < BLK) {
      const int r = tid;
      float mx = -1.0e30f;
      for (int j = 0; j < BLK; ++j) mx = fmaxf(mx, lS[r * LDS_S + j]);
      const float nm  = fmaxf(rowMax[r], mx);
      const float fac = __expf(rowMax[r] - nm);
      float s = 0.0f;
      for (int j = 0; j < BLK; ++j) {
        const float p = __expf(lS[r * LDS_S + j] - nm);
        lP[r * LDK + j] = f2bf(p);
        s += p;
      }
      rowSum[r] = rowSum[r] * fac + s;
      rowMax[r] = nm;
      rowFac[r] = fac;
    }
    __syncthreads();

    // ---- acc = acc*fac + P x V ----
    {
      v8f c0, c1;
      const int fA = (nh * 2) * 16 + l16;
      const int fB = (nh * 2 + 1) * 16 + l16;
#pragma unroll
      for (int i = 0; i < 8; ++i) {
        const int m = mt * 16 + i + (half ? 8 : 0);
        const float f = rowFac[m];
        c0[i] = lAcc[m * LDACC + fA] * f;
        c1[i] = lAcc[m * LDACC + fB] * f;
      }
#pragma unroll
      for (int kt = 0; kt < 2; ++kt) {
        const __bf16* pr = &lP[(mt * 16 + l16) * LDK + kt * 32 + (half ? 8 : 0)];
        v16bf a = cat8(ldv8(pr), ldv8(pr + 16));
        const int kk = kt * 32 + (half ? 8 : 0);
        const __bf16* p0 = &lVt[fA * LDK + kk];
        const __bf16* p1 = &lVt[fB * LDK + kk];
        v16bf b0 = cat8(ldv8(p0), ldv8(p0 + 16));
        v16bf b1 = cat8(ldv8(p1), ldv8(p1 + 16));
        c0 = wmma_bf16(a, b0, c0);
        c1 = wmma_bf16(a, b1, c1);
      }
#pragma unroll
      for (int i = 0; i < 8; ++i) {
        const int m = mt * 16 + i + (half ? 8 : 0);
        lAcc[m * LDACC + fA] = c0[i];
        lAcc[m * LDACC + fB] = c1[i];
      }
    }
    __syncthreads();
  }

  // ---- normalize and write bf16 head slice ----
#pragma unroll
  for (int e = 0; e < 16; ++e) {
    const float v = lAcc[s_row * LDACC + s_f + e] / rowSum[s_row];
    Out[(q_row0 + s_row) * D_MODEL + hcol + s_f + e] = f2bf(v);
  }
}

// ---------------- launcher ----------------
extern "C" void kernel_launch(void* const* d_in, const int* in_sizes, int n_in,
                              void* d_out, int out_size, void* d_ws, size_t ws_size,
                              hipStream_t stream) {
  (void)n_in; (void)out_size; (void)ws_size;
  const float* Q    = (const float*)d_in[0];
  const float* K    = (const float*)d_in[1];
  const float* V    = (const float*)d_in[2];
  const float* G    = (const float*)d_in[3];
  const float* mask = (const float*)d_in[4];
  const float* Wq   = (const float*)d_in[5];
  const float* bq   = (const float*)d_in[6];
  const float* Wk   = (const float*)d_in[7];
  const float* bk   = (const float*)d_in[8];
  const float* Wv   = (const float*)d_in[9];
  const float* bv   = (const float*)d_in[10];
  const float* Wo   = (const float*)d_in[11];
  const float* bo   = (const float*)d_in[12];

  const int Bb = in_sizes[0] / (SEQ * D_MODEL);   // batch (2)
  const size_t WN = (size_t)D_MODEL * D_MODEL;
  const size_t SROWS = (size_t)Bb * SEQ;
  const size_t GROWS = (size_t)Bb * G_LEN;

  size_t off = 0;
  char* base = (char*)d_ws;
  auto take = [&](size_t bytes) -> void* {
    void* p = base + off;
    off += (bytes + 255) & ~(size_t)255;
    return p;
  };
  __bf16* Wqt   = (__bf16*)take(WN * sizeof(__bf16));
  __bf16* Wkt   = (__bf16*)take(WN * sizeof(__bf16));
  __bf16* Wvt   = (__bf16*)take(WN * sizeof(__bf16));
  __bf16* Wot   = (__bf16*)take(WN * sizeof(__bf16));
  __bf16* Qp    = (__bf16*)take(SROWS * D_MODEL * sizeof(__bf16));
  __bf16* Kp    = (__bf16*)take(SROWS * D_MODEL * sizeof(__bf16));
  __bf16* Vp    = (__bf16*)take(SROWS * D_MODEL * sizeof(__bf16));
  __bf16* Gq    = (__bf16*)take(GROWS * D_MODEL * sizeof(__bf16));
  __bf16* Gkb   = (__bf16*)take(GROWS * D_MODEL * sizeof(__bf16));
  __bf16* Gvb   = (__bf16*)take(GROWS * D_MODEL * sizeof(__bf16));
  __bf16* Lattn = (__bf16*)take(SROWS * D_MODEL * sizeof(__bf16));
  __bf16* Gattn = (__bf16*)take(GROWS * D_MODEL * sizeof(__bf16));

  // 1) weights -> bf16, transposed to [N][K] for contiguous B-fragments
  dim3 gT(D_MODEL / 32, D_MODEL / 32);
  transpose_w_kernel<<<gT, 256, 0, stream>>>(Wq, Wqt);
  transpose_w_kernel<<<gT, 256, 0, stream>>>(Wk, Wkt);
  transpose_w_kernel<<<gT, 256, 0, stream>>>(Wv, Wvt);
  transpose_w_kernel<<<gT, 256, 0, stream>>>(Wo, Wot);

  // 2) projections (Q scaled by 1/sqrt(dk) at projection time, like reference)
  dim3 gBig((unsigned)(SROWS / BLK), D_MODEL / BLK);
  dim3 gSml((unsigned)(GROWS / BLK), D_MODEL / BLK);
  gemm_bias_kernel<true><<<gBig, 256, 0, stream>>>(Q, nullptr, Wqt, bq, SCALE_F, Qp, nullptr);
  gemm_bias_kernel<true><<<gBig, 256, 0, stream>>>(K, nullptr, Wkt, bk, 1.0f,    Kp, nullptr);
  gemm_bias_kernel<true><<<gBig, 256, 0, stream>>>(V, nullptr, Wvt, bv, 1.0f,    Vp, nullptr);
  gemm_bias_kernel<true><<<gSml, 256, 0, stream>>>(G, nullptr, Wqt, bq, SCALE_F, Gq,  nullptr);
  gemm_bias_kernel<true><<<gSml, 256, 0, stream>>>(G, nullptr, Wkt, bk, 1.0f,    Gkb, nullptr);
  gemm_bias_kernel<true><<<gSml, 256, 0, stream>>>(G, nullptr, Wvt, bv, 1.0f,    Gvb, nullptr);

  // 3) local attention: chunk0 = global K/V, chunk1 = own block
  attn_kernel<<<dim3(N_BLKS, N_HEADS, Bb), 256, 0, stream>>>(
      Qp, Kp, Vp, Gkb, Gvb, mask, Lattn, 2, 1.0f, 1);
  // 4) global attention: chunk0 = global K/V, chunks 1..128 = all local blocks
  //    (extra SCALE on scores reproduces the reference's double scaling)
  attn_kernel<<<dim3(1, N_HEADS, Bb), 256, 0, stream>>>(
      Gq, Kp, Vp, Gkb, Gvb, mask, Gattn, 1 + N_BLKS, SCALE_F, 0);

  // 5) output projections -> fp32 into d_out (local_out then global_out)
  float* out_local  = (float*)d_out;
  float* out_global = out_local + SROWS * D_MODEL;
  gemm_bias_kernel<false><<<gBig, 256, 0, stream>>>(nullptr, Lattn, Wot, bo, 1.0f, nullptr, out_local);
  gemm_bias_kernel<false><<<gSml, 256, 0, stream>>>(nullptr, Gattn, Wot, bo, 1.0f, nullptr, out_global);
}
